// LIFConvNet_50706383896824
// MI455X (gfx1250) — compile-verified, matrix operand order checked by reading, not run on hardware
//
#include <hip/hip_runtime.h>
#include <hip/hip_bf16.h>

typedef __attribute__((ext_vector_type(16))) _Float16 v16h;
typedef __attribute__((ext_vector_type(8)))  float    v8f;

#define DTM 0.1f    // DT * TAU_MEM_INV
#define DTS 0.2f    // DT * TAU_SYN_INV
#define VTH 0.2f

// ---------------------------------------------------------------------------
// f16 WMMA GEMM with implicit im2col and fused LIF epilogue.
//   MODE 0: dense B (Bf16 = [K x N] row-major), plain store to C (FC layer)
//   MODE 1: implicit im2col from f16 spike map Bf16 = [CIN][B*H*W], fused LIF
//   MODE 2: implicit im2col from f32 input  Bf32 = [B][CIN][H][W], fused LIF
// C/v/i/z are channel-major [M][N], N = B*H*W. K%64==0, N%32==0, M%16==0.
// Block = 256 threads (8 waves). Block tile: 128(M) x 32(N), K-chunk 64.
// Each wave: 16x32 of C -> 2 accumulators, 4 WMMAs per K-chunk.
// LDS tiles in *fragment order* (lane's 16 halfs contiguous -> 2x ds_load_b128):
//   A frag (s,w): halfs[((s*8+w)*32 + lane)*16 + elem]
//   B frag (s,j): halfs[((s*2+j)*32 + lane)*16 + elem]
// For K-position kl (0..31) of subchunk s:
//   half = (kl>>3)&1, p = ((kl&7)>>1) + (kl>=16?4:0), elem = 2p + (kl&1),
//   lane = (row_or_col & 15) + 16*half       (CDNA5 16-bit A/B VGPR layout)
// ---------------------------------------------------------------------------
template <int MODE, int CIN, int H, int W>
__global__ __launch_bounds__(256)
void gemm_conv_wmma(const _Float16* __restrict__ A,
                    const _Float16* __restrict__ Bf16,
                    const float* __restrict__ Bf32,
                    float* __restrict__ C,
                    float* __restrict__ v, float* __restrict__ i_,
                    _Float16* __restrict__ z,
                    int M, int N, int K)
{
    __shared__ _Float16 As[2 * 8 * 32 * 16];   // 16 KB, fragment order
    __shared__ _Float16 Bs[2 * 2 * 32 * 16];   //  4 KB, fragment order

    const int tid     = threadIdx.x;
    const int wave    = tid >> 5;
    const int lane    = tid & 31;
    const int n0      = blockIdx.x * 32;
    const int m_block = blockIdx.y * 128;
    const int m0      = m_block + wave * 16;

    v8f acc0 = {}, acc1 = {};

    for (int k0 = 0; k0 < K; k0 += 64) {
        // Prefetch next A K-tile (speculative, non-faulting).
        if (k0 + 64 < K) {
            int pr = m_block + (tid & 127);
            if (pr < M)
                __builtin_prefetch(A + (size_t)pr * K + k0 + 64, 0, 1);
            if constexpr (MODE == 0)
                __builtin_prefetch(Bf16 + (size_t)(k0 + 64 + (tid & 63)) * N + n0, 0, 1);
        }

        // ---- stage A tile 128 x 64 halfs = 4096 dwords, swizzled ----
        for (int idx = tid; idx < 4096; idx += 256) {
            int r = idx >> 5;            // tile row 0..127
            int d = idx & 31;            // dword within row, k = 2d
            int k = 2 * d;
            int s  = k >> 5;
            int kl = k & 31;
            int half = (kl >> 3) & 1;
            int pp   = ((kl & 7) >> 1) + ((kl >= 16) ? 4 : 0);
            int dl   = (r & 15) + (half << 4);
            int w    = r >> 4;
            unsigned val = 0u;
            int gm = m_block + r;
            if (gm < M)
                val = *(const unsigned*)(A + (size_t)gm * K + k0 + k);
            ((unsigned*)As)[((s * 8 + w) * 32 + dl) * 8 + pp] = val;
        }
        // ---- stage B tile 64 x 32 halfs, swizzled, implicit im2col ----
        for (int idx = tid; idx < 2048; idx += 256) {
            int k = idx >> 5;            // 0..63
            int n = idx & 31;            // 0..31
            int s  = k >> 5;
            int kl = k & 31;
            int half = (kl >> 3) & 1;
            int pp   = ((kl & 7) >> 1) + ((kl >= 16) ? 4 : 0);
            int j    = n >> 4;
            int dl   = (n & 15) + (half << 4);
            int elem = 2 * pp + (kl & 1);

            _Float16 val = (_Float16)0.f;
            if constexpr (MODE == 0) {
                val = Bf16[(size_t)(k0 + k) * N + n0 + n];
            } else {
                int kk = k0 + k;                      // im2col row (may be pad)
                if (kk < CIN * 9) {
                    int ci = kk / 9, rr = kk % 9, ky = rr / 3, kx = rr % 3;
                    int nn = n0 + n;                  // output pixel index
                    int xg = nn % W, yg = (nn / W) % H, bg = nn / (W * H);
                    int yy = yg + ky - 1, xx = xg + kx - 1;
                    if (yy >= 0 && yy < H && xx >= 0 && xx < W) {
                        if constexpr (MODE == 2)
                            val = (_Float16)Bf32[(((size_t)bg * CIN + ci) * H + yy) * W + xx];
                        else
                            val = Bf16[(size_t)ci * N + ((size_t)bg * H + yy) * W + xx];
                    }
                }
            }
            Bs[((s * 2 + j) * 32 + dl) * 16 + elem] = val;
        }
        __syncthreads();

        if (m0 < M) {   // wave-uniform: EXEC all-ones around WMMA
#pragma unroll
            for (int s = 0; s < 2; ++s) {
                v16h af = *(const v16h*)(As + ((s * 8 + wave) * 32 + lane) * 16);
                v16h b0 = *(const v16h*)(Bs + ((s * 2 + 0) * 32 + lane) * 16);
                v16h b1 = *(const v16h*)(Bs + ((s * 2 + 1) * 32 + lane) * 16);
                acc0 = __builtin_amdgcn_wmma_f32_16x16x32_f16(
                    false, af, false, b0, (short)0, acc0, false, false);
                acc1 = __builtin_amdgcn_wmma_f32_16x16x32_f16(
                    false, af, false, b1, (short)0, acc1, false, false);
            }
        }
        __syncthreads();
    }

    if (m0 < M) {
        // CDNA5 32-bit C/D layout: vgpr r -> M = r + 8*(lane>=16), N = lane&15
        const int nc = lane & 15;
        const int mr = (lane >> 4) * 8;
#pragma unroll
        for (int j = 0; j < 2; ++j) {
            const v8f* accp = j ? &acc1 : &acc0;
#pragma unroll
            for (int r = 0; r < 8; ++r) {
                size_t idx = (size_t)(m0 + mr + r) * N + n0 + j * 16 + nc;
                float inp = (*accp)[r];
                if constexpr (MODE == 0) {
                    C[idx] = inp;
                } else {
                    float vv = v[idx], ii = i_[idx];
                    float vdec = vv + DTM * ((0.f - vv) + ii);
                    float idec = ii * (1.f - DTS);
                    float zz = (vdec - VTH) > 0.f ? 1.f : 0.f;
                    v[idx]  = (1.f - zz) * vdec;      // V_RESET = 0
                    i_[idx] = idec + inp;
                    z[idx]  = (_Float16)zz;
                }
            }
        }
    }
}

// ---------------------------------------------------------------------------
// Weight prepack: f32 [Mreal x Kreal] -> f16 [M x Kp], zero padded.
// ---------------------------------------------------------------------------
__global__ void wcvt_kernel(const float* __restrict__ w, _Float16* __restrict__ a,
                            int M, int Mreal, int Kreal, int Kp)
{
    long long total = (long long)M * Kp;
    long long t = (long long)blockIdx.x * blockDim.x + threadIdx.x;
    if (t >= total) return;
    int m = (int)(t / Kp), k = (int)(t % Kp);
    float v = (m < Mreal && k < Kreal) ? w[(size_t)m * Kreal + k] : 0.f;
    a[t] = (_Float16)v;
}

// ---------------------------------------------------------------------------
// 2x2 maxpool on f16 channel-major [C][B*H*W] -> [C][B*(H/2)*(W/2)].
// ---------------------------------------------------------------------------
template <int H, int W>
__global__ void maxpool2_kernel(const _Float16* __restrict__ in, _Float16* __restrict__ out,
                                int C, int B)
{
    const int Ho = H / 2, Wo = W / 2;
    const int Ni = B * H * W, No = B * Ho * Wo;
    long long total = (long long)C * No;
    long long t = (long long)blockIdx.x * blockDim.x + threadIdx.x;
    if (t >= total) return;
    int c = (int)(t / No), n = (int)(t % No);
    int xo = n % Wo, yo = (n / Wo) % Ho, b = n / (Wo * Ho);
    const _Float16* base = in + (size_t)c * Ni + ((size_t)b * H + 2 * yo) * W + 2 * xo;
    float m = (float)base[0];
    m = fmaxf(m, (float)base[1]);
    m = fmaxf(m, (float)base[W]);
    m = fmaxf(m, (float)base[W + 1]);
    out[t] = (_Float16)m;
}

// ---------------------------------------------------------------------------
// Rearrange pooled z4 [256][B*16] -> FC B-matrix [4096][B] (K x N).
// feature f = c*16 + s matches reshape(B,-1) of [B,256,4,4].
// ---------------------------------------------------------------------------
__global__ void fc_rearrange_kernel(const _Float16* __restrict__ pooled,
                                    _Float16* __restrict__ bfc, int B)
{
    int t = blockIdx.x * blockDim.x + threadIdx.x;   // 4096*B
    if (t >= 4096 * B) return;
    int f = t / B, b = t % B;
    int c = f / 16, s = f % 16;
    bfc[(size_t)f * B + b] = pooled[(size_t)c * (B * 16) + b * 16 + s];
}

// ---------------------------------------------------------------------------
// LI step + voltage max over time. fc = [16][B] (rows = out, cols = batch).
// ---------------------------------------------------------------------------
__global__ void li_max_kernel(const float* __restrict__ fc,
                              const float* __restrict__ noise_t,
                              float* __restrict__ vl, float* __restrict__ il,
                              float* __restrict__ vmax, float* __restrict__ out,
                              int B, int t, int T)
{
    int idx = blockIdx.x * blockDim.x + threadIdx.x;  // b*10 + o
    if (idx >= B * 10) return;
    int b = idx / 10, o = idx % 10;
    float inp = fc[(size_t)o * B + b];
    float v = vl[idx], i = il[idx];
    float vn = v + DTM * ((0.f - v) + i);
    float in_ = i * (1.f - DTS) + inp;
    vl[idx] = vn; il[idx] = in_;
    float volt = vn + 0.001f * noise_t[idx];
    float m = (t == 0) ? volt : fmaxf(vmax[idx], volt);
    vmax[idx] = m;
    if (t == T - 1) out[idx] = m;
}

// ---------------------------------------------------------------------------
// Host side
// ---------------------------------------------------------------------------
static inline unsigned cdiv(long long a, long long b) { return (unsigned)((a + b - 1) / b); }

extern "C" void kernel_launch(void* const* d_in, const int* in_sizes, int n_in,
                              void* d_out, int out_size, void* d_ws, size_t ws_size,
                              hipStream_t stream)
{
    (void)in_sizes; (void)n_in; (void)out_size; (void)ws_size;
    const float* x     = (const float*)d_in[0];   // [16,32,3,32,32]
    const float* noise = (const float*)d_in[1];   // [16,32,10]
    const float* w1    = (const float*)d_in[2];   // [64,3,3,3]
    const float* w2    = (const float*)d_in[3];   // [128,64,3,3]
    const float* w3    = (const float*)d_in[4];   // [256,128,3,3]
    const float* w4    = (const float*)d_in[5];   // [256,256,3,3]
    const float* wfc   = (const float*)d_in[6];   // [10,4096]
    float* out = (float*)d_out;

    const int T = 16, B = 32;
    const int N1 = B * 32 * 32;   // 32768
    const int N3 = B * 16 * 16;   // 8192
    const int N4 = B * 8 * 8;     // 2048

    char* p = (char*)d_ws;
    auto alloc = [&](size_t bytes) -> void* {
        void* r = (void*)p;
        p += (bytes + 255) & ~(size_t)255;
        return r;
    };
    float*    v1   = (float*)alloc((size_t)64 * N1 * 4);
    float*    i1   = (float*)alloc((size_t)64 * N1 * 4);
    float*    v2   = (float*)alloc((size_t)128 * N1 * 4);
    float*    i2   = (float*)alloc((size_t)128 * N1 * 4);
    float*    v3   = (float*)alloc((size_t)256 * N3 * 4);
    float*    i3   = (float*)alloc((size_t)256 * N3 * 4);
    float*    v4   = (float*)alloc((size_t)256 * N4 * 4);
    float*    i4   = (float*)alloc((size_t)256 * N4 * 4);
    float*    vl   = (float*)alloc(320 * 4);
    float*    il   = (float*)alloc(320 * 4);
    float*    vmax = (float*)alloc(320 * 4);
    _Float16* z1   = (_Float16*)alloc((size_t)64 * N1 * 2);
    _Float16* z2   = (_Float16*)alloc((size_t)128 * N1 * 2);
    _Float16* z2p  = (_Float16*)alloc((size_t)128 * N3 * 2);
    _Float16* z3   = (_Float16*)alloc((size_t)256 * N3 * 2);
    _Float16* z3p  = (_Float16*)alloc((size_t)256 * N4 * 2);
    _Float16* z4   = (_Float16*)alloc((size_t)256 * N4 * 2);
    _Float16* z4p  = (_Float16*)alloc((size_t)256 * (B * 16) * 2);
    _Float16* a1   = (_Float16*)alloc((size_t)64 * 64 * 2);
    _Float16* a2   = (_Float16*)alloc((size_t)128 * 576 * 2);
    _Float16* a3   = (_Float16*)alloc((size_t)256 * 1152 * 2);
    _Float16* a4   = (_Float16*)alloc((size_t)256 * 2304 * 2);
    _Float16* afc  = (_Float16*)alloc((size_t)16 * 4096 * 2);
    _Float16* bfc  = (_Float16*)alloc((size_t)4096 * B * 2);
    float*    fco  = (float*)alloc((size_t)16 * B * 4);

    // Zero LIF/LI states (graph-capture safe).
    hipMemsetAsync(v1, 0, (size_t)64 * N1 * 4, stream);
    hipMemsetAsync(i1, 0, (size_t)64 * N1 * 4, stream);
    hipMemsetAsync(v2, 0, (size_t)128 * N1 * 4, stream);
    hipMemsetAsync(i2, 0, (size_t)128 * N1 * 4, stream);
    hipMemsetAsync(v3, 0, (size_t)256 * N3 * 4, stream);
    hipMemsetAsync(i3, 0, (size_t)256 * N3 * 4, stream);
    hipMemsetAsync(v4, 0, (size_t)256 * N4 * 4, stream);
    hipMemsetAsync(i4, 0, (size_t)256 * N4 * 4, stream);
    hipMemsetAsync(vl, 0, 320 * 4, stream);
    hipMemsetAsync(il, 0, 320 * 4, stream);

    // Prepack weights to f16 A matrices (K padded to %64).
    wcvt_kernel<<<cdiv((long long)64 * 64, 256), 256, 0, stream>>>(w1, a1, 64, 64, 27, 64);
    wcvt_kernel<<<cdiv((long long)128 * 576, 256), 256, 0, stream>>>(w2, a2, 128, 128, 576, 576);
    wcvt_kernel<<<cdiv((long long)256 * 1152, 256), 256, 0, stream>>>(w3, a3, 256, 256, 1152, 1152);
    wcvt_kernel<<<cdiv((long long)256 * 2304, 256), 256, 0, stream>>>(w4, a4, 256, 256, 2304, 2304);
    wcvt_kernel<<<cdiv((long long)16 * 4096, 256), 256, 0, stream>>>(wfc, afc, 16, 10, 4096, 4096);

    for (int t = 0; t < T; ++t) {
        const float* xt = x + (size_t)t * B * 3 * 1024;

        // layer 1: conv 3->64 on 32x32 (K pad 27->64), implicit im2col (f32 in), fused LIF
        gemm_conv_wmma<2, 3, 32, 32><<<dim3(N1 / 32, 1), 256, 0, stream>>>(
            a1, nullptr, xt, nullptr, v1, i1, z1, 64, N1, 64);

        // layer 2: conv 64->128 on 32x32, implicit im2col (f16 spikes), fused LIF
        gemm_conv_wmma<1, 64, 32, 32><<<dim3(N1 / 32, 1), 256, 0, stream>>>(
            a2, z1, nullptr, nullptr, v2, i2, z2, 128, N1, 576);
        maxpool2_kernel<32, 32><<<cdiv((long long)128 * N3, 256), 256, 0, stream>>>(z2, z2p, 128, B);

        // layer 3: conv 128->256 on 16x16, implicit im2col, fused LIF
        gemm_conv_wmma<1, 128, 16, 16><<<dim3(N3 / 32, 2), 256, 0, stream>>>(
            a3, z2p, nullptr, nullptr, v3, i3, z3, 256, N3, 1152);
        maxpool2_kernel<16, 16><<<cdiv((long long)256 * N4, 256), 256, 0, stream>>>(z3, z3p, 256, B);

        // layer 4: conv 256->256 on 8x8, implicit im2col, fused LIF
        gemm_conv_wmma<1, 256, 8, 8><<<dim3(N4 / 32, 2), 256, 0, stream>>>(
            a4, z3p, nullptr, nullptr, v4, i4, z4, 256, N4, 2304);
        maxpool2_kernel<8, 8><<<cdiv((long long)256 * B * 16, 256), 256, 0, stream>>>(z4, z4p, 256, B);

        // FC 4096->10 (padded to 16) + LI + time-max
        fc_rearrange_kernel<<<cdiv((long long)4096 * B, 256), 256, 0, stream>>>(z4p, bfc, B);
        gemm_conv_wmma<0, 0, 1, 1><<<dim3(1, 1), 256, 0, stream>>>(
            afc, bfc, nullptr, fco, nullptr, nullptr, nullptr, 16, B, 4096);
        li_max_kernel<<<2, 256, 0, stream>>>(fco, noise + (size_t)t * B * 10,
                                             vl, il, vmax, out, B, t, T);
    }
}
